// MoEReference_3813930959266
// MI455X (gfx1250) — compile-verified, matrix operand order checked by reading, not run on hardware
//
#include <hip/hip_runtime.h>

// ---------------------------------------------------------------------------
// MoE forward (Mixtral-style) for MI455X / gfx1250, wave32 + WMMA bf16.
// d_out layout (floats): [out T*H][topk_idx T*2][topk_w T*2][router_logits T*8]
// d_ws layout: [scale T*E f32][x_bf16 T*H]
// ---------------------------------------------------------------------------

#define T_TOK 8192
#define HID   1024
#define FFN   3584
#define NEXP  8
#define TOPK  2

typedef __bf16 bf16_t;
typedef __attribute__((ext_vector_type(16))) __bf16 v16bf;
typedef __attribute__((ext_vector_type(8)))  __bf16 v8bf;
typedef __attribute__((ext_vector_type(4)))  __bf16 v4bf;
typedef __attribute__((ext_vector_type(8)))  float  v8f;

__device__ __forceinline__ v16bf bf_combine(v8bf a, v8bf b) {
  v16bf f;
#pragma unroll
  for (int i = 0; i < 8; ++i) { f[i] = a[i]; f[i + 8] = b[i]; }
  return f;
}

// A fragment (16x32 bf16, M x K) from LDS, ISA layout:
// lanes 0-15: row = lane, K chunks {0-7,16-23}; lanes 16-31: row = lane-16,
// K chunks {8-15,24-31}.  Two 16-byte LDS loads per lane.
__device__ __forceinline__ v16bf afrag_lds(const bf16_t* base, int stride,
                                           int m0, int k0, int lane) {
  const int r  = lane & 15;
  const int kh = (lane >> 4) << 3;  // 0 or 8
  const bf16_t* p = base + (m0 + r) * stride + k0 + kh;
  return bf_combine(*(const v8bf*)p, *(const v8bf*)(p + 16));
}

// B fragment (32x16 bf16, K x N) built from an fp32 row-major weight matrix
// (row = output feature n, contiguous along K).  On-the-fly f32->bf16.
__device__ __forceinline__ v16bf bfrag_gbl_f32(const float* __restrict__ w,
                                               int ld, int n0, int k0, int lane) {
  const int n  = lane & 15;
  const int kh = (lane >> 4) << 3;
  const float* p = w + (size_t)(n0 + n) * ld + k0 + kh;
  v16bf f;
#pragma unroll
  for (int i = 0; i < 8; ++i) f[i] = (bf16_t)p[i];
#pragma unroll
  for (int i = 0; i < 8; ++i) f[i + 8] = (bf16_t)p[16 + i];
  return f;
}

__device__ __forceinline__ v8f wmma_bf16(v16bf a, v16bf b, v8f c) {
  return __builtin_amdgcn_wmma_f32_16x16x32_bf16(
      /*neg_a=*/false, a, /*neg_b=*/false, b,
      /*c_mod=*/(short)0, c, /*reuse_a=*/false, /*reuse_b=*/false);
}

// Async direct global->LDS copy of one 16-byte chunk (gfx1250 TDM-lite path,
// tracked with ASYNCcnt; no VGPR round-trip).
__device__ __forceinline__ void async_g2l_b128(uint32_t lds_addr,
                                               const void* gptr) {
  asm volatile("global_load_async_to_lds_b128 %0, %1, off"
               :: "v"(lds_addr), "v"((uint64_t)(uintptr_t)gptr)
               : "memory");
}
__device__ __forceinline__ void wait_asynccnt0() {
  asm volatile("s_wait_asynccnt 0x0" ::: "memory");
}

// ---------------------------------------------------------------------------
// Kernel 1: zero the accumulated output region (atomics accumulate into it).
// ---------------------------------------------------------------------------
__global__ __launch_bounds__(256) void moe_zero_out(float* __restrict__ out) {
  const size_t i = ((size_t)blockIdx.x * blockDim.x + threadIdx.x) * 4;
  float4 z = {0.f, 0.f, 0.f, 0.f};
  *(float4*)(out + i) = z;
}

// ---------------------------------------------------------------------------
// Kernel 2: convert hidden_states f32 -> bf16 (activation operand of WMMA).
// ---------------------------------------------------------------------------
__global__ __launch_bounds__(256) void moe_convert_x(const float* __restrict__ x,
                                                     bf16_t* __restrict__ xb) {
  const size_t i = ((size_t)blockIdx.x * blockDim.x + threadIdx.x) * 4;
  float4 v = *(const float4*)(x + i);
  v4bf o;
  o[0] = (bf16_t)v.x; o[1] = (bf16_t)v.y; o[2] = (bf16_t)v.z; o[3] = (bf16_t)v.w;
  *(v4bf*)(xb + i) = o;
}

// ---------------------------------------------------------------------------
// Kernel 3: router.  One wave32 per token; LDS-cached router weights.
// ---------------------------------------------------------------------------
__global__ __launch_bounds__(256) void moe_router(const float* __restrict__ x,
                                                  const float* __restrict__ rw,
                                                  float* __restrict__ out,
                                                  float* __restrict__ scale) {
  __shared__ float rws[NEXP * HID];  // 32 KB
  for (int i = threadIdx.x; i < NEXP * HID / 4; i += 256)
    ((float4*)rws)[i] = ((const float4*)rw)[i];
  __syncthreads();

  const int wave = threadIdx.x >> 5, lane = threadIdx.x & 31;
  const int t = blockIdx.x * 8 + wave;
  const float* xr = x + (size_t)t * HID;

  float p[NEXP];
#pragma unroll
  for (int e = 0; e < NEXP; ++e) p[e] = 0.f;
  for (int h = lane; h < HID; h += 32) {
    const float xv = xr[h];
#pragma unroll
    for (int e = 0; e < NEXP; ++e) p[e] += xv * rws[e * HID + h];
  }
#pragma unroll
  for (int off = 16; off > 0; off >>= 1)
#pragma unroll
    for (int e = 0; e < NEXP; ++e) p[e] += __shfl_xor(p[e], off, 32);

  if (lane == 0) {
    float mx = p[0];
#pragma unroll
    for (int e = 1; e < NEXP; ++e) mx = fmaxf(mx, p[e]);
    float sc[NEXP], s = 0.f;
#pragma unroll
    for (int e = 0; e < NEXP; ++e) { sc[e] = __expf(p[e] - mx); s += sc[e]; }
    const float inv = 1.f / s;
#pragma unroll
    for (int e = 0; e < NEXP; ++e) sc[e] *= inv;

    int i0 = 0;
#pragma unroll
    for (int e = 1; e < NEXP; ++e) if (sc[e] > sc[i0]) i0 = e;
    int i1 = (i0 == 0) ? 1 : 0;
#pragma unroll
    for (int e = 0; e < NEXP; ++e) if (e != i0 && sc[e] > sc[i1]) i1 = e;

    const size_t OFF_IDX = (size_t)T_TOK * HID;
    const size_t OFF_W   = OFF_IDX + (size_t)T_TOK * TOPK;
    const size_t OFF_LOG = OFF_W + (size_t)T_TOK * TOPK;
#pragma unroll
    for (int e = 0; e < NEXP; ++e) out[OFF_LOG + (size_t)t * NEXP + e] = p[e];
    out[OFF_IDX + (size_t)t * TOPK + 0] = (float)i0;
    out[OFF_IDX + (size_t)t * TOPK + 1] = (float)i1;
    out[OFF_W + (size_t)t * TOPK + 0] = sc[i0];
    out[OFF_W + (size_t)t * TOPK + 1] = sc[i1];
    float* srow = scale + (size_t)t * NEXP;
#pragma unroll
    for (int e = 0; e < NEXP; ++e)
      srow[e] = (e == i0) ? sc[i0] : ((e == i1) ? sc[i1] : 0.f);
  }
}

// ---------------------------------------------------------------------------
// Kernel 4: fused expert FFN:  out += scale[t][e] * (silu(x Wg^T) * (x Wu^T)) Wd^T
// Grid: (T/64 token tiles, NEXP experts).  Block: 512 threads = 16 waves.
// Per FFN super-chunk of 256:
//   phase 1: wave w computes gate/up for its 16 FFN rows (64x16 each) with
//            bf16 WMMA (K=1024 over async-staged LDS x chunks), silu*up -> hs.
//   phase 2: wave w owns hidden slice [w*64, w*64+64) and accumulates the
//            down projection over K=256 into 16 persistent f32 WMMA tiles.
// ---------------------------------------------------------------------------
#define TT   64                 // tokens per workgroup
#define SC   256                // FFN super-chunk (16 waves * 16 rows)
#define HSP  (SC + 8)           // hs row stride (keeps 16B alignment)

__global__ __launch_bounds__(512)
void moe_ffn(const bf16_t* __restrict__ xb,
             const float* __restrict__ wg, const float* __restrict__ wu,
             const float* __restrict__ wd, const float* __restrict__ scale,
             float* __restrict__ out) {
  __shared__ bf16_t xs[TT * 64];    // 8 KB  : x K-chunk stage (async-filled)
  __shared__ bf16_t hs[TT * HSP];   // 33 KB : silu(g)*u super-chunk

  const int e    = blockIdx.y;
  const int t0   = blockIdx.x * TT;
  const int wave = threadIdx.x >> 5;
  const int lane = threadIdx.x & 31;

  const float* wge = wg + (size_t)e * FFN * HID;
  const float* wue = wu + (size_t)e * FFN * HID;
  const float* wde = wd + (size_t)e * HID * FFN;

  v8f acc[4][4];
#pragma unroll
  for (int mt = 0; mt < 4; ++mt)
#pragma unroll
    for (int nt = 0; nt < 4; ++nt) acc[mt][nt] = (v8f)(0.f);

  const int n0 = wave * 64;  // this wave's hidden output slice

  // Per-thread 16B chunk of the 64x64 x stage.
  const int xrow  = threadIdx.x >> 3;
  const int xcol8 = (threadIdx.x & 7) * 8;
  const uint32_t xs_lds = (uint32_t)(uintptr_t)&xs[xrow * 64 + xcol8];
  const bf16_t* xg_base = xb + (size_t)(t0 + xrow) * HID + xcol8;

  for (int fc = 0; fc < FFN; fc += SC) {
    // ---------------- phase 1: gate/up for this wave's 16 FFN rows ----------
    const int fbase = fc + wave * 16;
    v8f gacc[4], uacc[4];
#pragma unroll
    for (int mt = 0; mt < 4; ++mt) { gacc[mt] = (v8f)(0.f); uacc[mt] = (v8f)(0.f); }

    // prefetch next super-chunk of this wave's down-proj weight slice
    if (fc + SC < FFN)
      __builtin_prefetch(wde + (size_t)(n0 + (lane & 15)) * FFN + fc + SC +
                             ((lane >> 4) << 3), 0, 1);

    for (int kb = 0; kb < HID; kb += 64) {
      __syncthreads();  // xs reuse / previous-phase completion
      // async stage: 64 tokens x 64 k of bf16 x directly into LDS
      async_g2l_b128(xs_lds, xg_base + kb);
      // prefetch next K-chunk of this wave's gate/up weight rows
      if (kb + 64 < HID) {
        __builtin_prefetch(wge + (size_t)(fbase + (lane & 15)) * HID + kb + 64, 0, 1);
        __builtin_prefetch(wue + (size_t)(fbase + (lane & 15)) * HID + kb + 64, 0, 1);
      }
      wait_asynccnt0();
      __syncthreads();
#pragma unroll
      for (int ks = 0; ks < 64; ks += 32) {
        const v16bf bg = bfrag_gbl_f32(wge, HID, fbase, kb + ks, lane);
        const v16bf bu = bfrag_gbl_f32(wue, HID, fbase, kb + ks, lane);
#pragma unroll
        for (int mt = 0; mt < 4; ++mt) {
          const v16bf a = afrag_lds(xs, 64, mt * 16, ks, lane);
          gacc[mt] = wmma_bf16(a, bg, gacc[mt]);
          uacc[mt] = wmma_bf16(a, bu, uacc[mt]);
        }
      }
    }

    // silu(g) * u -> hs (bf16).  C layout: VGPR r -> M = r + 8*(lane>=16),
    // N = lane&15.
#pragma unroll
    for (int mt = 0; mt < 4; ++mt)
#pragma unroll
      for (int r = 0; r < 8; ++r) {
        const float g = gacc[mt][r];
        const float h = (g / (1.f + __expf(-g))) * uacc[mt][r];
        const int m = mt * 16 + r + ((lane >> 4) << 3);
        hs[m * HSP + wave * 16 + (lane & 15)] = (bf16_t)h;
      }
    __syncthreads();

    // ---------------- phase 2: down projection over this super-chunk -------
#pragma unroll
    for (int ks = 0; ks < SC; ks += 32) {
      v16bf af[4];
#pragma unroll
      for (int mt = 0; mt < 4; ++mt) af[mt] = afrag_lds(hs, HSP, mt * 16, ks, lane);
#pragma unroll
      for (int nt = 0; nt < 4; ++nt) {
        const v16bf b = bfrag_gbl_f32(wde, FFN, n0 + nt * 16, fc + ks, lane);
#pragma unroll
        for (int mt = 0; mt < 4; ++mt)
          acc[mt][nt] = wmma_bf16(af[mt], b, acc[mt][nt]);
      }
    }
  }

  // epilogue: weighted accumulate into shared output across experts
#pragma unroll
  for (int mt = 0; mt < 4; ++mt)
#pragma unroll
    for (int r = 0; r < 8; ++r) {
      const int m = mt * 16 + r + ((lane >> 4) << 3);
      const float s = scale[(size_t)(t0 + m) * NEXP + e];
      if (s != 0.f) {
#pragma unroll
        for (int nt = 0; nt < 4; ++nt)
          atomicAdd(&out[(size_t)(t0 + m) * HID + n0 + nt * 16 + (lane & 15)],
                    s * acc[mt][nt][r]);
      }
    }
}

// ---------------------------------------------------------------------------
extern "C" void kernel_launch(void* const* d_in, const int* in_sizes, int n_in,
                              void* d_out, int out_size, void* d_ws, size_t ws_size,
                              hipStream_t stream) {
  (void)in_sizes; (void)n_in; (void)out_size; (void)ws_size;
  const float* x  = (const float*)d_in[0];
  const float* rw = (const float*)d_in[1];
  const float* wg = (const float*)d_in[2];
  const float* wu = (const float*)d_in[3];
  const float* wd = (const float*)d_in[4];
  float* out = (float*)d_out;

  float*  scale = (float*)d_ws;                                    // T*E f32
  bf16_t* xb = (bf16_t*)((char*)d_ws + (size_t)T_TOK * NEXP * 4);  // T*H bf16

  moe_zero_out<<<(T_TOK * HID) / (256 * 4), 256, 0, stream>>>(out);
  moe_convert_x<<<(T_TOK * HID) / (256 * 4), 256, 0, stream>>>(x, xb);
  moe_router<<<T_TOK / 8, 256, 0, stream>>>(x, rw, out, scale);
  moe_ffn<<<dim3(T_TOK / TT, NEXP), 512, 0, stream>>>(xb, wg, wu, wd, scale, out);
}
